// RSSM_19138374271049
// MI455X (gfx1250) — matrix operand order repaired and checked by reading, not compiled, not run on hardware
//
#include <hip/hip_runtime.h>
#include <stdint.h>

// ---------------------------------------------------------------------------
// RSSM scan for MI455X (gfx1250).
//  - 16 persistent workgroups, each owns a 16-row batch tile for all T steps.
//  - bf16 WMMA (v_wmma_f32_16x16x32_bf16) with f32 accumulation.
//  - Weights packed once into B-fragment order (bf16) in workspace; streamed
//    from L2 every step (8.8 MB fp32 -> 4.4 MB bf16 stays L2-resident).
//  - All recurrent state + activations in LDS (~102 KB/WG).
//  - 2 waves/SIMD -> amdgpu_waves_per_eu(2,2): 512 VGPRs/wave budget;
//    k-loops unrolled x2 with global_prefetch_b8 at distance 2.
// ---------------------------------------------------------------------------

namespace {

constexpr int   kT      = 256;
constexpr int   kOutW   = 1216;
constexpr float kEps    = 1e-3f;
constexpr float kMinStd = 0.1f;
constexpr float kUpdB   = -1.0f;

typedef __attribute__((ext_vector_type(16))) __bf16 v16bf;
typedef __attribute__((ext_vector_type(8)))  float  v8f;

union Frag { v16bf v; uint4 q[2]; };

__device__ __forceinline__ unsigned short f2bfbits(float x) {
  unsigned u = __float_as_uint(x);
  u += 0x7FFFu + ((u >> 16) & 1u);          // round-to-nearest-even-ish
  return (unsigned short)(u >> 16);
}
__device__ __forceinline__ float sigm(float x)  { return 1.f / (1.f + __expf(-x)); }
__device__ __forceinline__ float silu(float x)  { return x / (1.f + __expf(-x)); }
__device__ __forceinline__ float softp(float x) { return (x > 20.f) ? x : logf(1.f + __expf(x)); }
__device__ __forceinline__ v8f   zero8() { v8f z = {0.f,0.f,0.f,0.f,0.f,0.f,0.f,0.f}; return z; }

// ds_swizzle xor-butterfly within 16-lane halves (VALU-free lane index).
// Group-of-32 encoding: offset = {xor_mask[14:10], or_mask[9:5], and_mask[4:0]}.
template<int OFF>
__device__ __forceinline__ float swz_xor(float x) {
  int i = __builtin_amdgcn_ds_swizzle(__float_as_int(x), (OFF << 10) | 0x1f);
  return __int_as_float(i);
}

// ---------------------------------------------------------------------------
// Wave-level GEMM: 16 rows x (NTI 16-col tiles), K = KT*32, bf16 WMMA.
// A: row-major bf16 in LDS (pitchA elements). A-fragment per §7.12.2:
//   lane l: row = l&15, q[0] = K (l>>4)*8 .. +7 ; q[1] = 16 + (l>>4)*8 .. +7
// B: pre-packed fragments, lane holds 16 contiguous bf16 (2 x b128).
// Tiles: base+0 .. base+NTI-1 (NTI <= 4 keeps register pressure bounded).
// unroll 2 + prefetch(dist 2) hides the L2 round trip of the weight stream.
// ---------------------------------------------------------------------------
template<int KT, int NTI>
__device__ __forceinline__ void wmma_gemm(const unsigned short* sA, int pitchA,
                                          const unsigned short* __restrict__ Bp,
                                          int NT, int base, v8f* acc, int lane) {
  const int row = lane & 15;
  const int h   = lane >> 4;
  const unsigned short* arow = sA + row * pitchA + h * 8;
  #pragma unroll 2
  for (int kt = 0; kt < KT; ++kt) {
    if (kt + 2 < KT) {
      #pragma unroll
      for (int i = 0; i < NTI; ++i)
        __builtin_prefetch(Bp + (size_t)((kt + 2) * NT + base + i) * 512 + lane * 16, 0, 1);
    }
    Frag a;
    const uint4* ap = (const uint4*)(arow + kt * 32);
    a.q[0] = ap[0];          // K = kt*32 + h*8 + 0..7
    a.q[1] = ap[2];          // K = kt*32 + h*8 + 16..23
    #pragma unroll
    for (int i = 0; i < NTI; ++i) {
      Frag b;
      const uint4* bp = (const uint4*)(Bp + (size_t)(kt * NT + base + i) * 512 + lane * 16);
      b.q[0] = bp[0];
      b.q[1] = bp[1];
      acc[i] = __builtin_amdgcn_wmma_f32_16x16x32_bf16(
          false, a.v, false, b.v, (short)0, acc[i], false, false);
    }
  }
}

// ---------------------------------------------------------------------------
// Row-wise LN reduction core. p[r]/q[r] are this lane's partial sum / sumsq
// for row r + 8*(lane>>4) over its columns (one per tile). Butterfly over the 16 lanes of
// each half via ds_swizzle with all shuffles of a step batched.
// ---------------------------------------------------------------------------
#define LN_BF_STEP(OFF)                                                        \
  {                                                                            \
    float tp[8], tq[8];                                                        \
    _Pragma("unroll")                                                          \
    for (int r = 0; r < 8; ++r) tp[r] = swz_xor<OFF>(p[r]);                    \
    _Pragma("unroll")                                                          \
    for (int r = 0; r < 8; ++r) tq[r] = swz_xor<OFF>(q[r]);                    \
    _Pragma("unroll")                                                          \
    for (int r = 0; r < 8; ++r) { p[r] += tp[r]; q[r] += tq[r]; }              \
  }

__device__ __forceinline__ void ln_reduce(float* p, float* q, float invN,
                                          int w, int lane, int tid,
                                          float (*sRed)[16][2],
                                          float* sMean, float* sRstd) {
  LN_BF_STEP(8)
  LN_BF_STEP(4)
  LN_BF_STEP(2)
  LN_BF_STEP(1)
  const int h = lane >> 4;
  if ((lane & 15) == 0) {
    #pragma unroll
    for (int r = 0; r < 8; ++r) { sRed[w][r + 8 * h][0] = p[r]; sRed[w][r + 8 * h][1] = q[r]; }
  }
  __syncthreads();
  if (tid < 16) {
    float S = 0.f, Q = 0.f;
    #pragma unroll
    for (int ww = 0; ww < 8; ++ww) { S += sRed[ww][tid][0]; Q += sRed[ww][tid][1]; }
    float m = S * invN;
    sMean[tid] = m;
    sRstd[tid] = rsqrtf(Q * invN - m * m + kEps);
  }
  __syncthreads();
}

template<int NTI>
__device__ __forceinline__ void ln_stats(const v8f* acc, float invN, int w, int lane,
                                         int tid, float (*sRed)[16][2],
                                         float* sMean, float* sRstd) {
  float p[8], q[8];
  #pragma unroll
  for (int r = 0; r < 8; ++r) {
    p[r] = 0.f; q[r] = 0.f;
    #pragma unroll
    for (int i = 0; i < NTI; ++i) { float x = acc[i][r]; p[r] += x; q[r] += x * x; }
  }
  ln_reduce(p, q, invN, w, lane, tid, sRed, sMean, sRstd);
}

__device__ __forceinline__ void ln_stats3(const v8f* a0, const v8f* a1, const v8f* a2,
                                          float invN, int w, int lane, int tid,
                                          float (*sRed)[16][2],
                                          float* sMean, float* sRstd) {
  float p[8], q[8];
  #pragma unroll
  for (int r = 0; r < 8; ++r) {
    p[r] = 0.f; q[r] = 0.f;
    #pragma unroll
    for (int i = 0; i < 4; ++i) {
      float x = a0[i][r]; p[r] += x; q[r] += x * x;
      float y = a1[i][r]; p[r] += y; q[r] += y * y;
      float z = a2[i][r]; p[r] += z; q[r] += z * z;
    }
  }
  ln_reduce(p, q, invN, w, lane, tid, sRed, sMean, sRstd);
}

// Apply LN + silu, emit bf16 into a row-major LDS A-buffer (4 tiles/wave).
__device__ __forceinline__ void apply_ln_silu4(const v8f* acc, int base,
                                               const float* __restrict__ g,
                                               const float* __restrict__ b,
                                               unsigned short* dest, int pitch, int lane,
                                               const float* sMean, const float* sRstd) {
  const int h = lane >> 4, c0 = lane & 15;
  #pragma unroll
  for (int i = 0; i < 4; ++i) {
    const int col = (base + i) * 16 + c0;
    const float gg = g[col], bb = b[col];
    #pragma unroll
    for (int r = 0; r < 8; ++r) {
      const int row = r + 8 * h;
      float x = (acc[i][r] - sMean[row]) * sRstd[row] * gg + bb;
      dest[row * pitch + col] = f2bfbits(silu(x));
    }
  }
}

// ---------------------------------------------------------------------------
// Prep kernel 1: pack fp32 [K][N] weights -> bf16 B-fragment order.
// dst[frag*512 + lane*16 + j] = W[kt*32 + (j>>3)*16 + (lane>>4)*8 + (j&7)][nt*16 + (lane&15)]
// with zero padding for K >= Kreal.
// ---------------------------------------------------------------------------
__global__ void pack_weights(const float* __restrict__ src, unsigned short* __restrict__ dst,
                             int Kreal, int Kp, int N) {
  int idx = blockIdx.x * 256 + threadIdx.x;
  int total = Kp * N;
  if (idx >= total) return;
  int f = idx >> 9, r = idx & 511;
  int lane = r >> 4, j = r & 15;
  int NT = N >> 4;
  int kt = f / NT, nt = f - kt * NT;
  int K = kt * 32 + (j >> 3) * 16 + (lane >> 4) * 8 + (j & 7);
  int n = nt * 16 + (lane & 15);
  float v = (K < Kreal) ? src[(size_t)K * N + n] : 0.f;
  dst[idx] = f2bfbits(v);
}

// ---------------------------------------------------------------------------
// Prep kernel 2: deter0 = tanh(initial_deter); init_mean from prior head.
// ---------------------------------------------------------------------------
__global__ void prep_init(const float* __restrict__ initial_deter,
                          const float* __restrict__ w_prior_out,
                          const float* __restrict__ g_po, const float* __restrict__ b_po,
                          const float* __restrict__ w_prior_stats,
                          const float* __restrict__ b_ps,
                          float* __restrict__ deter0_out, float* __restrict__ initm_out) {
  __shared__ float sd[512], sh[512], wp[16], wq[16], stat[2];
  int tid = threadIdx.x;
  float d = tanhf(initial_deter[tid]);
  sd[tid] = d;
  deter0_out[tid] = d;
  __syncthreads();
  float accv = 0.f;
  for (int k = 0; k < 512; ++k) accv += sd[k] * w_prior_out[k * 512 + tid];
  float p = accv, q = accv * accv;
  for (int off = 16; off; off >>= 1) { p += __shfl_xor(p, off, 32); q += __shfl_xor(q, off, 32); }
  if ((tid & 31) == 0) { wp[tid >> 5] = p; wq[tid >> 5] = q; }
  __syncthreads();
  if (tid == 0) {
    float S = 0.f, Q = 0.f;
    for (int i = 0; i < 16; ++i) { S += wp[i]; Q += wq[i]; }
    float m = S / 512.f;
    stat[0] = m;
    stat[1] = rsqrtf(Q / 512.f - m * m + kEps);
  }
  __syncthreads();
  float x = (accv - stat[0]) * stat[1] * g_po[tid] + b_po[tid];
  sh[tid] = silu(x);
  __syncthreads();
  if (tid < 32) {
    float a2 = 0.f;
    for (int k = 0; k < 512; ++k) a2 += sh[k] * w_prior_stats[k * 64 + tid];
    initm_out[tid] = a2 + b_ps[tid];
  }
}

// ---------------------------------------------------------------------------
// Main persistent kernel: 16 blocks x 256 threads, each block = 16 batch rows.
// ---------------------------------------------------------------------------
__global__ __launch_bounds__(256)
__attribute__((amdgpu_waves_per_eu(2, 2)))
void rssm_scan(
    const float* __restrict__ obs, const float* __restrict__ action,
    const int* __restrict__ isf,
    const float* __restrict__ g1, const float* __restrict__ b1,
    const float* __restrict__ gg, const float* __restrict__ bg,
    const float* __restrict__ go, const float* __restrict__ bo,
    const float* __restrict__ bps,
    const float* __restrict__ gp, const float* __restrict__ bp_,
    const float* __restrict__ bqs,
    const unsigned short* __restrict__ W1p, const unsigned short* __restrict__ Wgp,
    const unsigned short* __restrict__ Wop, const unsigned short* __restrict__ Wsp,
    const unsigned short* __restrict__ Wpp, const unsigned short* __restrict__ Wqp,
    const float* __restrict__ deter0c, const float* __restrict__ initm,
    float* __restrict__ out) {

  __shared__ __align__(16) unsigned short sA_x1[16][64];     // [stoch|act|pad] bf16
  __shared__ __align__(16) unsigned short sA_hd[16][1024];   // [h | deter] bf16
  __shared__ __align__(16) unsigned short sA_po[16][544];    // [deter|obs|pad] bf16
  __shared__ __align__(16) unsigned short sA_hs[16][512];    // h2 / h3 bf16
  __shared__ float sDeter[16][512];
  __shared__ float sStoch[16][32];
  __shared__ float sRed[8][16][2];
  __shared__ float sMean[16], sRstd[16];
  __shared__ int   sMask[16];

  const int tid  = threadIdx.x;
  const int w    = tid >> 5;
  const int lane = tid & 31;
  const int m0   = blockIdx.x * 16;
  const int h    = lane >> 4;
  const int c0   = lane & 15;

  // Initial carry + constant pads.
  for (int e = tid; e < 16 * 512; e += 256) { int row = e >> 9, j = e & 511; sDeter[row][j] = deter0c[j]; }
  for (int e = tid; e < 16 * 32;  e += 256) { int row = e >> 5, j = e & 31;  sStoch[row][j] = initm[j]; }
  for (int e = tid; e < 16 * 16;  e += 256) { int row = e >> 4, j = e & 15;  if (j < 14) sA_po[row][530 + j] = 0; }
  __syncthreads();

  for (int t = 0; t < kT; ++t) {
    // ---- step start: reset mask + build bf16 GEMM inputs --------------------
    if (tid < 16) sMask[tid] = (isf[(size_t)(m0 + tid) * kT + t] > 0);
    __syncthreads();
    for (int e = tid; e < 16 * 512; e += 256) {
      int row = e >> 9, j = e & 511;
      if (sMask[row]) sDeter[row][j] = deter0c[j];
    }
    for (int e = tid; e < 16 * 32; e += 256) {
      int row = e >> 5, j = e & 31;
      if (sMask[row]) sStoch[row][j] = initm[j];
    }
    __syncthreads();
    for (int e = tid; e < 16 * 512; e += 256) {
      int row = e >> 9, j = e & 511;
      sA_hd[row][512 + j] = f2bfbits(sDeter[row][j]);
    }
    for (int e = tid; e < 16 * 64; e += 256) {
      int row = e >> 6, j = e & 63;
      float v = 0.f;
      if (j < 32)      v = sStoch[row][j];
      else if (j < 36) v = sMask[row] ? 0.f : action[((size_t)(m0 + row) * kT + t) * 4 + (j - 32)];
      sA_x1[row][j] = f2bfbits(v);
    }
    for (int e = tid; e < 16 * 32; e += 256) {
      int row = e >> 5, j = e & 31;
      if (j < 18) sA_po[row][512 + j] = f2bfbits(obs[((size_t)(m0 + row) * kT + t) * 18 + j]);
    }
    __syncthreads();

    // ---- stage 1: h = silu(LN(x1 @ W_prior_in)) ----------------------------
    {
      v8f acc[4]; for (int i = 0; i < 4; ++i) acc[i] = zero8();
      wmma_gemm<2, 4>(&sA_x1[0][0], 64, W1p, 32, 4 * w, acc, lane);
      ln_stats<4>(acc, 1.f / 512.f, w, lane, tid, sRed, sMean, sRstd);
      apply_ln_silu4(acc, 4 * w, g1, b1, &sA_hd[0][0], 1024, lane, sMean, sRstd);
    }
    __syncthreads();

    // ---- stage 2: GRU gates = LN([h|deter] @ W_gru); 3 passes (r/c/u) ------
    {
      v8f aR[4], aC[4], aU[4];
      for (int i = 0; i < 4; ++i) { aR[i] = zero8(); aC[i] = zero8(); aU[i] = zero8(); }
      wmma_gemm<32, 4>(&sA_hd[0][0], 1024, Wgp, 96,      4 * w, aR, lane);   // r gates
      wmma_gemm<32, 4>(&sA_hd[0][0], 1024, Wgp, 96, 32 + 4 * w, aC, lane);   // c gates
      wmma_gemm<32, 4>(&sA_hd[0][0], 1024, Wgp, 96, 64 + 4 * w, aU, lane);   // u gates
      ln_stats3(aR, aC, aU, 1.f / 1536.f, w, lane, tid, sRed, sMean, sRstd);
      #pragma unroll
      for (int i = 0; i < 4; ++i) {
        const int j = 64 * w + 16 * i + c0;
        const float gr = gg[j],        br = bg[j];
        const float gc = gg[512 + j],  bc = bg[512 + j];
        const float gu = gg[1024 + j], bu = bg[1024 + j];
        #pragma unroll
        for (int r = 0; r < 8; ++r) {
          const int row = r + 8 * h;
          const float m = sMean[row], rs = sRstd[row];
          float rr   = sigm((aR[i][r] - m) * rs * gr + br);
          float cc   =      (aC[i][r] - m) * rs * gc + bc;
          float uu   = sigm((aU[i][r] - m) * rs * gu + bu + kUpdB);
          float cand = silu(rr * cc);
          float nd   = uu * cand + (1.f - uu) * sDeter[row][j];
          sDeter[row][j]      = nd;
          sA_po[row][j]       = f2bfbits(nd);
          sA_hd[row][512 + j] = f2bfbits(nd);
          size_t ob = ((size_t)(m0 + row) * kT + t) * kOutW;
          out[ob + j]       = nd;     // post deter
          out[ob + 608 + j] = nd;     // prior deter
        }
      }
    }
    __syncthreads();

    // ---- stage 4: h2 = silu(LN(deter @ W_prior_out)) -----------------------
    {
      v8f acc[4]; for (int i = 0; i < 4; ++i) acc[i] = zero8();
      wmma_gemm<16, 4>(&sA_hd[0][512], 1024, Wop, 32, 4 * w, acc, lane);
      ln_stats<4>(acc, 1.f / 512.f, w, lane, tid, sRed, sMean, sRstd);
      apply_ln_silu4(acc, 4 * w, go, bo, &sA_hs[0][0], 512, lane, sMean, sRstd);
    }
    __syncthreads();

    // ---- stage 5: prior stats (waves 0..3) ---------------------------------
    if (w < 4) {
      v8f acc[1]; acc[0] = zero8();
      wmma_gemm<16, 1>(&sA_hs[0][0], 512, Wsp, 4, w, acc, lane);
      const int col = w * 16 + c0;
      const float bias = bps[col];
      #pragma unroll
      for (int r = 0; r < 8; ++r) {
        const int row = r + 8 * h;
        float x = acc[0][r] + bias;
        size_t ob = ((size_t)(m0 + row) * kT + t) * kOutW;
        if (col < 32) { out[ob + 1120 + col] = x; out[ob + 1152 + col] = x; }
        else          { out[ob + 1184 + (col - 32)] = softp(x) + kMinStd; }
      }
    }

    // ---- stage 6: h3 = silu(LN([deter|obs] @ W_post)) ----------------------
    {
      v8f acc[4]; for (int i = 0; i < 4; ++i) acc[i] = zero8();
      wmma_gemm<17, 4>(&sA_po[0][0], 544, Wpp, 32, 4 * w, acc, lane);
      ln_stats<4>(acc, 1.f / 512.f, w, lane, tid, sRed, sMean, sRstd);
      apply_ln_silu4(acc, 4 * w, gp, bp_, &sA_hs[0][0], 512, lane, sMean, sRstd);
    }
    __syncthreads();

    // ---- stage 7: posterior stats (waves 0..3); stoch carry = qm -----------
    if (w < 4) {
      v8f acc[1]; acc[0] = zero8();
      wmma_gemm<16, 1>(&sA_hs[0][0], 512, Wqp, 4, w, acc, lane);
      const int col = w * 16 + c0;
      const float bias = bqs[col];
      #pragma unroll
      for (int r = 0; r < 8; ++r) {
        const int row = r + 8 * h;
        float x = acc[0][r] + bias;
        size_t ob = ((size_t)(m0 + row) * kT + t) * kOutW;
        if (col < 32) { out[ob + 512 + col] = x; out[ob + 544 + col] = x; sStoch[row][col] = x; }
        else          { out[ob + 576 + (col - 32)] = softp(x) + kMinStd; }
      }
    }
    __syncthreads();
  }
}

}  // namespace

// ---------------------------------------------------------------------------
// Workspace layout (bytes):
//   W1p @ 0        : 64   x 512  bf16 =   65536
//   Wgp @ 65536    : 1024 x 1536 bf16 = 3145728
//   Wop @ 3211264  : 512  x 512  bf16 =  524288
//   Wsp @ 3735552  : 512  x 64   bf16 =   65536
//   Wpp @ 3801088  : 544  x 512  bf16 =  557056
//   Wqp @ 4358144  : 512  x 64   bf16 =   65536
//   deter0 @ 4423680 : 512 f32 ; init_mean @ 4425728 : 32 f32
// ---------------------------------------------------------------------------
extern "C" void kernel_launch(void* const* d_in, const int* in_sizes, int n_in,
                              void* d_out, int out_size, void* d_ws, size_t ws_size,
                              hipStream_t stream) {
  (void)in_sizes; (void)n_in; (void)out_size; (void)ws_size;
  const float* obs           = (const float*)d_in[0];
  const float* action        = (const float*)d_in[1];
  const int*   is_first      = (const int*)d_in[2];
  const float* w_prior_in    = (const float*)d_in[3];
  const float* g_prior_in    = (const float*)d_in[4];
  const float* bb_prior_in   = (const float*)d_in[5];
  const float* w_gru         = (const float*)d_in[6];
  const float* g_gru         = (const float*)d_in[7];
  const float* bb_gru        = (const float*)d_in[8];
  const float* w_prior_out   = (const float*)d_in[9];
  const float* g_prior_out   = (const float*)d_in[10];
  const float* bb_prior_out  = (const float*)d_in[11];
  const float* w_prior_stats = (const float*)d_in[12];
  const float* b_prior_stats = (const float*)d_in[13];
  const float* w_post        = (const float*)d_in[14];
  const float* g_post        = (const float*)d_in[15];
  const float* bb_post       = (const float*)d_in[16];
  const float* w_post_stats  = (const float*)d_in[17];
  const float* b_post_stats  = (const float*)d_in[18];
  const float* initial_deter = (const float*)d_in[19];
  float* out = (float*)d_out;
  char*  ws  = (char*)d_ws;

  unsigned short* W1p = (unsigned short*)(ws + 0);
  unsigned short* Wgp = (unsigned short*)(ws + 65536);
  unsigned short* Wop = (unsigned short*)(ws + 3211264);
  unsigned short* Wsp = (unsigned short*)(ws + 3735552);
  unsigned short* Wpp = (unsigned short*)(ws + 3801088);
  unsigned short* Wqp = (unsigned short*)(ws + 4358144);
  float* deter0c = (float*)(ws + 4423680);
  float* initm   = (float*)(ws + 4425728);

  pack_weights<<<  128, 256, 0, stream>>>(w_prior_in,    W1p,   36,   64,  512);
  pack_weights<<< 6144, 256, 0, stream>>>(w_gru,         Wgp, 1024, 1024, 1536);
  pack_weights<<< 1024, 256, 0, stream>>>(w_prior_out,   Wop,  512,  512,  512);
  pack_weights<<<  128, 256, 0, stream>>>(w_prior_stats, Wsp,  512,  512,   64);
  pack_weights<<< 1088, 256, 0, stream>>>(w_post,        Wpp,  530,  544,  512);
  pack_weights<<<  128, 256, 0, stream>>>(w_post_stats,  Wqp,  512,  512,   64);

  prep_init<<<1, 512, 0, stream>>>(initial_deter, w_prior_out, g_prior_out,
                                   bb_prior_out, w_prior_stats, b_prior_stats,
                                   deter0c, initm);

  rssm_scan<<<16, 256, 0, stream>>>(
      obs, action, is_first,
      g_prior_in, bb_prior_in, g_gru, bb_gru, g_prior_out, bb_prior_out,
      b_prior_stats, g_post, bb_post, b_post_stats,
      W1p, Wgp, Wop, Wsp, Wpp, Wqp,
      deter0c, initm, out);
}
